// LSTM_84396107366491
// MI455X (gfx1250) — matrix operand order repaired and checked by reading, not compiled, not run on hardware
//
#include <hip/hip_runtime.h>
#include <hip/hip_bf16.h>

// ---- problem dims ----
#define V_DIM 10000
#define E_DIM 512
#define H_DIM 512
#define B_DIM 64
#define T_DIM 128
#define KCAT  1536          // E + H (x part) + H (h part)
#define NGATES 2048         // 4*H
#define KT_CAT 48           // KCAT/32
#define NT_CAT 128          // NGATES/16
#define MT 4                // B/16
#define KT_OUT 16           // H/32
#define NT_OUT 625          // V/16 (10000 = 625*16 exactly)

typedef __attribute__((ext_vector_type(16))) __bf16 v16bf;
typedef __attribute__((ext_vector_type(8)))  float  v8f;

__device__ __forceinline__ __bf16 f2bf(float f) {
  unsigned u = __builtin_bit_cast(unsigned, f);
  unsigned r = (u + 0x7FFFu + ((u >> 16) & 1u)) >> 16;   // RNE
  unsigned short s = (unsigned short)r;
  return __builtin_bit_cast(__bf16, s);
}

// ISA 16-bit A-matrix 16x32 layout: given row m (0..15) and k_local (0..31),
// return element index e (0..15) and lane (0..31).
__device__ __forceinline__ int a_lane_elem(int m, int kl, int* lane) {
  bool hi = (kl & 8) != 0;                 // k in 8..15 or 24..31 -> lanes 16..31
  *lane = hi ? (m + 16) : m;
  int e;
  if (!hi) e = (kl < 16) ? kl : (kl - 8);  // lanes 0..15 hold K 0..7, 16..23
  else     e = (kl < 16) ? (kl - 8) : (kl - 16); // lanes 16..31 hold K 8..15, 24..31
  return e;
}

// C/D 16x16 f32 fragment store: VGPR r -> M = r + (lane<16 ? 0 : 8), N = lane%16.
__device__ __forceinline__ void store_ctile(float* __restrict__ dst, size_t rowStride,
                                            int mbase, int nbase, const v8f& acc,
                                            float bv) {
#pragma unroll
  for (int r = 0; r < 8; ++r)
    dst[(size_t)(mbase + r) * rowStride + nbase] = acc[r] + bv;
}

// ---------------- one-time packing kernels ----------------

// W_cat = [W_ih (2048x1024) | W_hh (2048x512)] packed as bf16 B-fragments.
// B-layout (16-bit, 32x16): N=lane%16, K=8*(lane/16)+(e&7)+(e>=8 ? 16 : 0).
__global__ __launch_bounds__(256) void pack_wcat(const float* __restrict__ W_ih,
                                                 const float* __restrict__ W_hh,
                                                 __bf16* __restrict__ dst) {
  int p = blockIdx.x * 256 + threadIdx.x;
  const int total = NT_CAT * KT_CAT * 32 * 16;
  if (p >= total) return;
  int e    = p & 15;
  int lane = (p >> 4) & 31;
  int kt   = (p >> 9) % KT_CAT;
  int nt   = p / (512 * KT_CAT);
  int n = nt * 16 + (lane & 15);
  int k = kt * 32 + (lane >> 4) * 8 + (e & 7) + ((e & 8) << 1);
  float v = (k < 1024) ? W_ih[n * 1024 + k] : W_hh[n * 512 + (k - 1024)];
  dst[p] = f2bf(v);
}

__global__ __launch_bounds__(256) void pack_wout(const float* __restrict__ W_out,
                                                 __bf16* __restrict__ dst) {
  int p = blockIdx.x * 256 + threadIdx.x;
  const int total = NT_OUT * KT_OUT * 32 * 16;
  if (p >= total) return;
  int e    = p & 15;
  int lane = (p >> 4) & 31;
  int kt   = (p >> 9) % KT_OUT;
  int nt   = p / (512 * KT_OUT);
  int n = nt * 16 + (lane & 15);
  int k = kt * 32 + (lane >> 4) * 8 + (e & 7) + ((e & 8) << 1);
  dst[p] = f2bf(W_out[n * 512 + k]);
}

__global__ __launch_bounds__(256) void fold_bias(const float* __restrict__ b_ih,
                                                 const float* __restrict__ b_hh,
                                                 float* __restrict__ bias) {
  int n = blockIdx.x * 256 + threadIdx.x;
  if (n < NGATES) bias[n] = b_ih[n] + b_hh[n];
}

// h0 = c0 = context_vec; x0 = [context, emb[labels[:,0]], h0] into packed A.
__global__ __launch_bounds__(256) void init_state(const float* __restrict__ ctx,
                                                  const int* __restrict__ labels,
                                                  const float* __restrict__ emb,
                                                  float* __restrict__ c,
                                                  __bf16* __restrict__ xcatA) {
  int idx = blockIdx.x * 256 + threadIdx.x;
  if (idx >= B_DIM * H_DIM) return;
  int b = idx >> 9, j = idx & 511;
  float cv = ctx[idx];
  c[idx] = cv;
  __bf16 cb = f2bf(cv);
  int mt = b >> 4, m = b & 15;
  int lane, e, kt, kl, K;
  // context portion, K = j
  kt = j >> 5; kl = j & 31; e = a_lane_elem(m, kl, &lane);
  xcatA[((mt * KT_CAT + kt) * 32 + lane) * 16 + e] = cb;
  // hidden portion, K = 1024 + j (h0 = context)
  K = 1024 + j; kt = K >> 5; kl = K & 31; e = a_lane_elem(m, kl, &lane);
  xcatA[((mt * KT_CAT + kt) * 32 + lane) * 16 + e] = cb;
  // input portion, K = 512 + j, from emb[labels[b,0]]
  int lab = labels[b * T_DIM + 0];
  K = 512 + j; kt = K >> 5; kl = K & 31; e = a_lane_elem(m, kl, &lane);
  xcatA[((mt * KT_CAT + kt) * 32 + lane) * 16 + e] = f2bf(emb[(size_t)lab * E_DIM + j]);
}

__global__ __launch_bounds__(256) void zero_t0(float* __restrict__ out) {
  int idx = blockIdx.x * 256 + threadIdx.x;
  if (idx >= B_DIM * V_DIM) return;
  int b = idx / V_DIM, v = idx % V_DIM;
  out[(size_t)b * (T_DIM * V_DIM) + v] = 0.0f;
}

// ---------------- per-step kernels ----------------

// gates = xcat @ Wcat.T + (b_ih+b_hh).
// 2x2 register tiling: each wave computes M-tiles {2*mtg, 2*mtg+1} x
// N-tiles {2*ntg, 2*ntg+1}; per K-step: 2 A-frag + 2 B-frag loads -> 4 WMMAs.
__global__ __launch_bounds__(256) void gates_gemm(const __bf16* __restrict__ Apack,
                                                  const __bf16* __restrict__ Bpack,
                                                  const float* __restrict__ bias,
                                                  float* __restrict__ gates) {
  int wave = threadIdx.x >> 5;
  int lane = threadIdx.x & 31;
  int tile = blockIdx.x * 8 + wave;      // 16 blocks * 8 waves = 128 wave-tiles
  int mtg = tile >> 6;                   // 0..1
  int ntg = tile & 63;                   // 0..63
  int mt0 = mtg * 2, mt1 = mt0 + 1;
  int nt0 = ntg * 2, nt1 = nt0 + 1;
  const v16bf* a0 = reinterpret_cast<const v16bf*>(Apack + ((size_t)(mt0 * KT_CAT) * 32 + lane) * 16);
  const v16bf* a1 = reinterpret_cast<const v16bf*>(Apack + ((size_t)(mt1 * KT_CAT) * 32 + lane) * 16);
  const v16bf* b0 = reinterpret_cast<const v16bf*>(Bpack + ((size_t)(nt0 * KT_CAT) * 32 + lane) * 16);
  const v16bf* b1 = reinterpret_cast<const v16bf*>(Bpack + ((size_t)(nt1 * KT_CAT) * 32 + lane) * 16);
  v8f acc00 = {}, acc01 = {}, acc10 = {}, acc11 = {};
  for (int kt = 0; kt < KT_CAT; ++kt) {
    v16bf A0 = a0[kt * 32];              // one kt tile = 512 bf16 = 32 v16bf
    v16bf A1 = a1[kt * 32];
    v16bf B0 = b0[kt * 32];
    v16bf B1 = b1[kt * 32];
    acc00 = __builtin_amdgcn_wmma_f32_16x16x32_bf16(false, A0, false, B0, (short)0, acc00, false, false);
    acc01 = __builtin_amdgcn_wmma_f32_16x16x32_bf16(false, A0, false, B1, (short)0, acc01, false, false);
    acc10 = __builtin_amdgcn_wmma_f32_16x16x32_bf16(false, A1, false, B0, (short)0, acc10, false, false);
    acc11 = __builtin_amdgcn_wmma_f32_16x16x32_bf16(false, A1, false, B1, (short)0, acc11, false, false);
  }
  int mlo = (lane < 16) ? 0 : 8;
  int nlo = lane & 15;
  float bv0 = bias[nt0 * 16 + nlo];
  float bv1 = bias[nt1 * 16 + nlo];
  store_ctile(gates, NGATES, mt0 * 16 + mlo, nt0 * 16 + nlo, acc00, bv0);
  store_ctile(gates, NGATES, mt0 * 16 + mlo, nt1 * 16 + nlo, acc01, bv1);
  store_ctile(gates, NGATES, mt1 * 16 + mlo, nt0 * 16 + nlo, acc10, bv0);
  store_ctile(gates, NGATES, mt1 * 16 + mlo, nt1 * 16 + nlo, acc11, bv1);
}

// LSTM nonlinearity; writes h2 as bf16 straight into both packed-A buffers.
__global__ __launch_bounds__(256) void lstm_cell(const float* __restrict__ gates,
                                                 float* __restrict__ c,
                                                 __bf16* __restrict__ xcatA,
                                                 __bf16* __restrict__ hA) {
  int idx = blockIdx.x * 256 + threadIdx.x;
  if (idx >= B_DIM * H_DIM) return;
  int b = idx >> 9, j = idx & 511;
  const float* g = gates + (size_t)b * NGATES;
  float ig = 1.0f / (1.0f + __expf(-g[j]));
  float fg = 1.0f / (1.0f + __expf(-g[j + 512]));
  float gg = tanhf(g[j + 1024]);
  float og = 1.0f / (1.0f + __expf(-g[j + 1536]));
  float c2 = fg * c[idx] + ig * gg;
  c[idx] = c2;
  __bf16 hb = f2bf(og * tanhf(c2));
  int mt = b >> 4, m = b & 15;
  int lane, e, kt, kl, K;
  kt = j >> 5; kl = j & 31; e = a_lane_elem(m, kl, &lane);
  hA[((mt * KT_OUT + kt) * 32 + lane) * 16 + e] = hb;
  K = 1024 + j; kt = K >> 5; kl = K & 31; e = a_lane_elem(m, kl, &lane);
  xcatA[((mt * KT_CAT + kt) * 32 + lane) * 16 + e] = hb;
}

// logits[:, t, :] = h2 @ W_out.T + b_out ; 2x2 register tiling.
// 313 N-groups (last group has one valid N tile), 2 M-groups -> 626 wave-tiles.
__global__ __launch_bounds__(256) void out_gemm(const __bf16* __restrict__ hA,
                                                const __bf16* __restrict__ WoutP,
                                                const float* __restrict__ b_out,
                                                float* __restrict__ out, int t) {
  const int NTG = (NT_OUT + 1) / 2;      // 313
  int wave = threadIdx.x >> 5;
  int lane = threadIdx.x & 31;
  int tile = blockIdx.x * 8 + wave;
  if (tile >= 2 * NTG) return;
  int mtg = tile / NTG;                  // 0..1
  int ntg = tile % NTG;                  // 0..312
  int mt0 = mtg * 2, mt1 = mt0 + 1;
  int nt0 = ntg * 2, nt1 = nt0 + 1;
  bool hasN1 = (nt1 < NT_OUT);
  const v16bf* a0 = reinterpret_cast<const v16bf*>(hA + ((size_t)(mt0 * KT_OUT) * 32 + lane) * 16);
  const v16bf* a1 = reinterpret_cast<const v16bf*>(hA + ((size_t)(mt1 * KT_OUT) * 32 + lane) * 16);
  const v16bf* b0 = reinterpret_cast<const v16bf*>(WoutP + ((size_t)(nt0 * KT_OUT) * 32 + lane) * 16);
  const v16bf* b1 = reinterpret_cast<const v16bf*>(WoutP + ((size_t)((hasN1 ? nt1 : nt0) * KT_OUT) * 32 + lane) * 16);
  v8f acc00 = {}, acc01 = {}, acc10 = {}, acc11 = {};
#pragma unroll
  for (int kt = 0; kt < KT_OUT; ++kt) {
    v16bf A0 = a0[kt * 32];
    v16bf A1 = a1[kt * 32];
    v16bf B0 = b0[kt * 32];
    v16bf B1 = b1[kt * 32];
    acc00 = __builtin_amdgcn_wmma_f32_16x16x32_bf16(false, A0, false, B0, (short)0, acc00, false, false);
    acc01 = __builtin_amdgcn_wmma_f32_16x16x32_bf16(false, A0, false, B1, (short)0, acc01, false, false);
    acc10 = __builtin_amdgcn_wmma_f32_16x16x32_bf16(false, A1, false, B0, (short)0, acc10, false, false);
    acc11 = __builtin_amdgcn_wmma_f32_16x16x32_bf16(false, A1, false, B1, (short)0, acc11, false, false);
  }
  float* outT = out + (size_t)t * V_DIM;
  const size_t rowStride = (size_t)T_DIM * V_DIM;
  int mlo = (lane < 16) ? 0 : 8;
  int nlo = lane & 15;
  float bv0 = b_out[nt0 * 16 + nlo];
  store_ctile(outT, rowStride, mt0 * 16 + mlo, nt0 * 16 + nlo, acc00, bv0);
  store_ctile(outT, rowStride, mt1 * 16 + mlo, nt0 * 16 + nlo, acc10, bv0);
  if (hasN1) {
    float bv1 = b_out[nt1 * 16 + nlo];
    store_ctile(outT, rowStride, mt0 * 16 + mlo, nt1 * 16 + nlo, acc01, bv1);
    store_ctile(outT, rowStride, mt1 * 16 + mlo, nt1 * 16 + nlo, acc11, bv1);
  }
}

// Per batch row: argmax over V logits (LDS reduce), pick label vs predicted
// embedding, scatter bf16 into the input slice (K = 512..1023) of packed A.
__global__ __launch_bounds__(256) void next_input(const float* __restrict__ out, int t,
                                                  const int* __restrict__ labels,
                                                  const unsigned char* __restrict__ use_label,
                                                  const float* __restrict__ emb,
                                                  __bf16* __restrict__ xcatA) {
  __shared__ float sv[256];
  __shared__ int   si[256];
  int b = blockIdx.x;
  int tid = threadIdx.x;
  const float* row = out + (size_t)b * (T_DIM * V_DIM) + (size_t)t * V_DIM;
  float best = -3.4e38f; int bi = 0;
  for (int v = tid; v < V_DIM; v += 256) {
    float x = row[v];
    if (x > best || (x == best && v < bi)) { best = x; bi = v; }
  }
  sv[tid] = best; si[tid] = bi;
  __syncthreads();
  for (int s = 128; s > 0; s >>= 1) {
    if (tid < s) {
      if (sv[tid + s] > sv[tid] ||
          (sv[tid + s] == sv[tid] && si[tid + s] < si[tid])) {
        sv[tid] = sv[tid + s]; si[tid] = si[tid + s];
      }
    }
    __syncthreads();
  }
  int src_row = use_label[t] ? labels[b * T_DIM + t] : si[0];
  int mt = b >> 4, m = b & 15;
  for (int j = tid; j < E_DIM; j += 256) {
    int K = 512 + j;
    int kt = K >> 5, kl = K & 31, lane;
    int e = a_lane_elem(m, kl, &lane);
    xcatA[((mt * KT_CAT + kt) * 32 + lane) * 16 + e] = f2bf(emb[(size_t)src_row * E_DIM + j]);
  }
}

// ---------------- host launch ----------------

extern "C" void kernel_launch(void* const* d_in, const int* in_sizes, int n_in,
                              void* d_out, int out_size, void* d_ws, size_t ws_size,
                              hipStream_t stream) {
  const float* ctx   = (const float*)d_in[0];
  const int*   labels= (const int*)d_in[1];
  const unsigned char* use_label = (const unsigned char*)d_in[2];
  const float* emb   = (const float*)d_in[3];
  const float* W_ih  = (const float*)d_in[4];
  const float* W_hh  = (const float*)d_in[5];
  const float* b_ih  = (const float*)d_in[6];
  const float* b_hh  = (const float*)d_in[7];
  const float* W_out = (const float*)d_in[8];
  const float* b_out = (const float*)d_in[9];
  float* out = (float*)d_out;

  char* ws = (char*)d_ws;
  size_t o = 0;
  auto carve = [&](size_t bytes) { size_t r = o; o += (bytes + 255) & ~(size_t)255; return r; };
  __bf16* WcatP = (__bf16*)(ws + carve((size_t)NGATES * KCAT * 2));      // 6.0 MB
  __bf16* WoutP = (__bf16*)(ws + carve((size_t)V_DIM * H_DIM * 2));      // 10.0 MB
  float*  bias  = (float*)(ws + carve((size_t)NGATES * 4));
  __bf16* xcatA = (__bf16*)(ws + carve((size_t)B_DIM * KCAT * 2));
  __bf16* hA    = (__bf16*)(ws + carve((size_t)B_DIM * H_DIM * 2));
  float*  gates = (float*)(ws + carve((size_t)B_DIM * NGATES * 4));
  float*  cbuf  = (float*)(ws + carve((size_t)B_DIM * H_DIM * 4));
  (void)ws_size; (void)in_sizes; (void)n_in; (void)out_size;

  // one-time packing + state init
  {
    int tot = NT_CAT * KT_CAT * 512;
    pack_wcat<<<(tot + 255) / 256, 256, 0, stream>>>(W_ih, W_hh, WcatP);
  }
  {
    int tot = NT_OUT * KT_OUT * 512;
    pack_wout<<<(tot + 255) / 256, 256, 0, stream>>>(W_out, WoutP);
  }
  fold_bias<<<(NGATES + 255) / 256, 256, 0, stream>>>(b_ih, b_hh, bias);
  init_state<<<(B_DIM * H_DIM + 255) / 256, 256, 0, stream>>>(ctx, labels, emb, cbuf, xcatA);
  zero_t0<<<(B_DIM * V_DIM + 255) / 256, 256, 0, stream>>>(out);

  const int gatesBlocks = (2 * (NT_CAT / 2)) / 8;            // 128 wave-tiles -> 16 blocks
  const int cellBlocks  = (B_DIM * H_DIM) / 256;             // 128
  const int outTiles    = 2 * ((NT_OUT + 1) / 2);            // 626 wave-tiles
  const int outBlocks   = (outTiles + 7) / 8;                // 79

  for (int t = 1; t < T_DIM; ++t) {
    gates_gemm<<<gatesBlocks, 256, 0, stream>>>(xcatA, WcatP, bias, gates);
    lstm_cell<<<cellBlocks, 256, 0, stream>>>(gates, cbuf, xcatA, hA);
    out_gemm<<<outBlocks, 256, 0, stream>>>(hA, WoutP, b_out, out, t);
    if (t + 1 < T_DIM)
      next_input<<<B_DIM, 256, 0, stream>>>(out, t, labels, use_label, emb, xcatA);
  }
}
